// BaseRGAT_12962211299980
// MI455X (gfx1250) — matrix operand order repaired and checked by reading, not compiled, not run on hardware
//
#include <hip/hip_runtime.h>
#include <hip/hip_bf16.h>
#include <cstdint>

// ---------------------------------------------------------------------------
// R-GAT for MI455X (gfx1250, wave32).
//  - Projection GEMM (x @ W[r]) in bf16 via v_wmma_f32_16x16x32_bf16,
//    operands pre-swizzled into the CDNA5 A/B register layouts so the
//    inner loop is pure b128 loads + WMMA.
//  - Edge phase (gather/softmax/scatter) in fp32 with f32/u32 global atomics;
//    this phase is the HBM-bound part (~4.2 GB total -> ~180us at 23.3 TB/s).
// ---------------------------------------------------------------------------

typedef __attribute__((ext_vector_type(16))) __bf16 v16bf;
typedef __attribute__((ext_vector_type(8)))  float  v8f;

#define IN_F   256
#define HEADS  4
#define DH     64
#define RELS   6

// ---- helpers ---------------------------------------------------------------

__device__ __forceinline__ uint16_t f32_to_bf16_rne(float f) {
  uint32_t u = __float_as_uint(f);
  uint32_t r = 0x7FFFu + ((u >> 16) & 1u);
  return (uint16_t)((u + r) >> 16);
}

// order-preserving float -> uint map (monotone under unsigned compare)
__device__ __forceinline__ uint32_t f2ord(float f) {
  uint32_t u = __float_as_uint(f);
  return (u & 0x80000000u) ? ~u : (u | 0x80000000u);
}
__device__ __forceinline__ float ord2f(uint32_t u) {
  uint32_t v = (u & 0x80000000u) ? (u & 0x7FFFFFFFu) : ~u;
  return __uint_as_float(v);
}

// ---- one-time pack of X into WMMA A-matrix panel layout --------------------
// A (16x32 bf16 per k-step): lane<16 -> rows m=lane, K-offsets {0..7,16..23};
// lane>=16 -> rows m=lane-16, K-offsets {8..15,24..31}.  dword j packs (k,k+1).
// Panel index: ((tm*8 + kstep)*32 + lane)*8 + j   (dwords)
__global__ void __launch_bounds__(256)
pack_x_bf16(const float* __restrict__ X, uint32_t* __restrict__ Xs, int total) {
  int idx = blockIdx.x * blockDim.x + threadIdx.x;
  if (idx >= total) return;
  int j     =  idx        & 7;
  int lane  = (idx >> 3)  & 31;
  int kstep = (idx >> 8)  & 7;
  int tm    =  idx >> 11;
  int hi    = (lane >= 16) ? 8 : 0;
  int koff  = (j < 4) ? (hi + 2 * j) : (16 + hi + 2 * (j - 4));
  int k     = kstep * 32 + koff;
  int m     = tm * 16 + (lane & 15);
  const float* p = X + (size_t)m * IN_F + k;
  Xs[idx] = (uint32_t)f32_to_bf16_rne(p[0]) |
            ((uint32_t)f32_to_bf16_rne(p[1]) << 16);
}

// ---- one-time pack of W into WMMA B-matrix panel layout --------------------
// B (32x16 bf16 per k-step): lane&15 = column n; lane<16 -> K {0..7,16..23},
// lane>=16 -> K {8..15,24..31}; dword j packs (k,k+1).
// Panel index: (((r*16 + tn)*8 + kstep)*32 + lane)*8 + j
__global__ void __launch_bounds__(256)
pack_w_bf16(const float* __restrict__ W, uint32_t* __restrict__ Ws, int total) {
  int idx = blockIdx.x * blockDim.x + threadIdx.x;
  if (idx >= total) return;
  int j     =  idx        & 7;
  int lane  = (idx >> 3)  & 31;
  int kstep = (idx >> 8)  & 7;
  int tn    = (idx >> 11) & 15;
  int r     =  idx >> 15;
  int hi    = (lane >= 16) ? 8 : 0;
  int k     = kstep * 32 + ((j < 4) ? (hi + 2 * j) : (16 + hi + 2 * (j - 4)));
  int n     = tn * 16 + (lane & 15);
  const float* base = W + (size_t)r * IN_F * IN_F;
  Ws[idx] = (uint32_t)f32_to_bf16_rne(base[(size_t)k * IN_F + n]) |
            ((uint32_t)f32_to_bf16_rne(base[(size_t)(k + 1) * IN_F + n]) << 16);
}

// ---- bf16 WMMA GEMM: feat[M,256] = X[M,256] @ W[256,256] -------------------
// One wave per 16x16 output tile, K=256 -> 8 x v_wmma_f32_16x16x32_bf16.
__global__ void __launch_bounds__(256)
rgat_gemm_wmma(const uint32_t* __restrict__ Xs, const uint32_t* __restrict__ Ws,
               float* __restrict__ feat, int Mtiles) {
  const int lane = threadIdx.x & 31;
  const int wave = (blockIdx.x * blockDim.x + threadIdx.x) >> 5;
  const int tm   = wave >> 4;       // 16 N-tiles (256/16)
  const int tn   = wave & 15;
  if (tm >= Mtiles) return;         // wave-uniform; grid sized exactly anyway

  v8f acc = {};
  const uint4* ap = (const uint4*)Xs + ((size_t)tm * 256 + lane) * 2;
  const uint4* bp = (const uint4*)Ws + ((size_t)tn * 256 + lane) * 2;
#pragma unroll
  for (int ks = 0; ks < 8; ++ks) {
    union { v16bf v; uint4 q[2]; } a, b;
    a.q[0] = ap[0]; a.q[1] = ap[1];
    b.q[0] = bp[0]; b.q[1] = bp[1];
    ap += 64; bp += 64;             // next k-step panel (32 lanes * 2 uint4)
    acc = __builtin_amdgcn_wmma_f32_16x16x32_bf16(
        /*neg_a=*/false, a.v, /*neg_b=*/false, b.v,
        /*c_mod=*/(short)0, acc, /*reuse_a=*/false, /*reuse_b=*/false);
  }
  // C/D layout: VGPR i -> row i (lanes 0-15) / row 8+i (lanes 16-31), col=lane&15
  const int col   = tn * 16 + (lane & 15);
  const int rbase = tm * 16 + ((lane >= 16) ? 8 : 0);
  float* out = feat + (size_t)rbase * IN_F + col;
#pragma unroll
  for (int i = 0; i < 8; ++i) out[(size_t)i * IN_F] = acc[i];
}

// ---- el/er: per-(node,head) dot of feat with attn vectors ------------------
__global__ void __launch_bounds__(256)
attn_dot(const float* __restrict__ feat, const float* __restrict__ al,
         const float* __restrict__ ar, float* __restrict__ el,
         float* __restrict__ er, int NH) {
  const int lane = threadIdx.x & 31;
  const int wave = (blockIdx.x * blockDim.x + threadIdx.x) >> 5;
  if (wave >= NH) return;
  const int n = wave >> 2, h = wave & 3;
  const float* f  = feat + (size_t)n * IN_F + h * DH;
  const float* vl = al + h * DH;
  const float* vr = ar + h * DH;
  float sl = f[lane] * vl[lane] + f[lane + 32] * vl[lane + 32];
  float sr = f[lane] * vr[lane] + f[lane + 32] * vr[lane + 32];
#pragma unroll
  for (int off = 16; off; off >>= 1) {
    sl += __shfl_xor(sl, off, 32);
    sr += __shfl_xor(sr, off, 32);
  }
  if (lane == 0) { el[wave] = sl; er[wave] = sr; }
}

// ---- per-relation init: zero agg, z, mkey ----------------------------------
__global__ void __launch_bounds__(256)
init_rel(float* __restrict__ agg, float* __restrict__ z,
         uint32_t* __restrict__ mkey, int totalAgg, int totalNH) {
  int idx = blockIdx.x * blockDim.x + threadIdx.x;
  if (idx < totalAgg) agg[idx] = 0.0f;
  if (idx < totalNH) { z[idx] = 0.0f; mkey[idx] = 0u; }  // 0 < f2ord(-inf)
}

__global__ void __launch_bounds__(256)
zero_f32(float* __restrict__ p, int n) {
  int idx = blockIdx.x * blockDim.x + threadIdx.x;
  if (idx < n) p[idx] = 0.0f;
}

// ---- edge pass 1: leaky-relu score, stash it, segment max via u32 atomics --
__global__ void __launch_bounds__(256)
edge_score_max(const int* __restrict__ src, const int* __restrict__ dst,
               const float* __restrict__ el, const float* __restrict__ er,
               uint32_t* __restrict__ mkey, float* __restrict__ escore, int EH) {
  int idx = blockIdx.x * blockDim.x + threadIdx.x;
  if (idx >= EH) return;
  int e = idx >> 2, h = idx & 3;
  int s = src[e], d = dst[e];
  float v = el[s * HEADS + h] + er[d * HEADS + h];
  v = (v > 0.0f) ? v : 0.2f * v;            // leaky_relu slope 0.2
  escore[idx] = v;
  atomicMax(&mkey[d * HEADS + h], f2ord(v));
}

// ---- edge pass 2: a = exp(e - m[d]); z[d] += a ------------------------------
__global__ void __launch_bounds__(256)
edge_exp_sum(const int* __restrict__ dst, const uint32_t* __restrict__ mkey,
             float* __restrict__ escore, float* __restrict__ z, int EH) {
  int idx = blockIdx.x * blockDim.x + threadIdx.x;
  if (idx >= EH) return;
  int e = idx >> 2, h = idx & 3;
  int d = dst[e];
  float m = ord2f(mkey[d * HEADS + h]);
  uint32_t mb = __float_as_uint(m);
  if ((mb & 0x7F800000u) == 0x7F800000u) m = 0.0f;  // non-finite -> 0
  float a = __expf(escore[idx] - m);
  escore[idx] = a;                                   // overwrite with exp value
  atomicAdd(&z[d * HEADS + h], a);
}

// ---- edge pass 3: agg[dst] += alpha * feat[src]  (wave per edge) -----------
__global__ void __launch_bounds__(256)
edge_aggregate(const int* __restrict__ src, const int* __restrict__ dst,
               const float* __restrict__ aexp, const float* __restrict__ z,
               const float* __restrict__ feat, float* __restrict__ agg, int E) {
  const int lane = threadIdx.x & 31;
  const int e    = (blockIdx.x * blockDim.x + threadIdx.x) >> 5;
  if (e >= E) return;
  const int s = src[e], d = dst[e];
  const int h = lane >> 3;                      // 8 lanes per head (64 ch)
  float alpha = aexp[e * HEADS + h] / fmaxf(z[d * HEADS + h], 1e-9f);
  const float4* fs = (const float4*)(feat + (size_t)s * IN_F + lane * 8);
  float*        ag = agg + (size_t)d * IN_F + lane * 8;
  float4 f0 = fs[0], f1 = fs[1];
  atomicAdd(ag + 0, alpha * f0.x); atomicAdd(ag + 1, alpha * f0.y);
  atomicAdd(ag + 2, alpha * f0.z); atomicAdd(ag + 3, alpha * f0.w);
  atomicAdd(ag + 4, alpha * f1.x); atomicAdd(ag + 5, alpha * f1.y);
  atomicAdd(ag + 6, alpha * f1.z); atomicAdd(ag + 7, alpha * f1.w);
}

// ---- out += relu(agg + bias[r]) --------------------------------------------
__global__ void __launch_bounds__(256)
accum_out(const float* __restrict__ agg, const float* __restrict__ bias,
          float* __restrict__ out, int total) {
  int idx = blockIdx.x * blockDim.x + threadIdx.x;
  if (idx >= total) return;
  out[idx] += fmaxf(agg[idx] + bias[idx & (IN_F - 1)], 0.0f);
}

__global__ void __launch_bounds__(256)
final_relu(float* __restrict__ out, int n) {
  int idx = blockIdx.x * blockDim.x + threadIdx.x;
  if (idx < n) out[idx] = fmaxf(out[idx], 0.0f);
}

// ---------------------------------------------------------------------------
extern "C" void kernel_launch(void* const* d_in, const int* in_sizes, int n_in,
                              void* d_out, int out_size, void* d_ws, size_t ws_size,
                              hipStream_t stream) {
  (void)n_in; (void)out_size; (void)ws_size;
  const float* x      = (const float*)d_in[0];   // [N,256]
  const float* W      = (const float*)d_in[1];   // [6,256,256]
  const float* attn_l = (const float*)d_in[2];   // [6,4,64]
  const float* attn_r = (const float*)d_in[3];   // [6,4,64]
  const float* bias   = (const float*)d_in[4];   // [6,256]
  const int*   src    = (const int*)d_in[5];     // [6,E]
  const int*   dst    = (const int*)d_in[6];     // [6,E]
  float*       out    = (float*)d_out;           // [N,256]

  const int N  = in_sizes[0] / IN_F;             // 50000
  const int E  = in_sizes[5] / RELS;             // 300000
  const int Mtiles = N / 16;                     // 3125 (exact)
  const int NC = N * IN_F;                       // node*channel count
  const int NH = N * HEADS;
  const int EH = E * HEADS;

  // workspace carve-out (256B aligned)
  char* wsp = (char*)d_ws;
  size_t off = 0;
  auto take = [&](size_t bytes) -> char* {
    char* p = wsp + off; off += (bytes + 255) & ~(size_t)255; return p;
  };
  uint32_t* Xs   = (uint32_t*)take((size_t)NC * 2);               // bf16 X, swizzled
  uint32_t* Wsz  = (uint32_t*)take((size_t)RELS * IN_F * IN_F * 2);
  float*    feat = (float*)take((size_t)NC * 4);
  float*    agg  = (float*)take((size_t)NC * 4);
  float*    el   = (float*)take((size_t)NH * 4);
  float*    er   = (float*)take((size_t)NH * 4);
  float*    z    = (float*)take((size_t)NH * 4);
  uint32_t* mkey = (uint32_t*)take((size_t)NH * 4);
  float*    aexp = (float*)take((size_t)EH * 4);

  const int B = 256;
  auto blk = [](int n, int b) { return (n + b - 1) / b; };

  // one-time bf16 conversion + WMMA-panel swizzle
  pack_x_bf16<<<blk(NC / 2, B), B, 0, stream>>>(x, Xs, NC / 2);
  const int wTot = RELS * (IN_F * IN_F / 2);
  pack_w_bf16<<<blk(wTot, B), B, 0, stream>>>(W, Wsz, wTot);
  zero_f32<<<blk(NC, B), B, 0, stream>>>(out, NC);

  for (int r = 0; r < RELS; ++r) {
    init_rel<<<blk(NC, B), B, 0, stream>>>(agg, z, mkey, NC, NH);
    rgat_gemm_wmma<<<blk(Mtiles * 16 * 32, B), B, 0, stream>>>(
        Xs, Wsz + (size_t)r * (IN_F * IN_F / 2), feat, Mtiles);
    attn_dot<<<blk(NH * 32, B), B, 0, stream>>>(
        feat, attn_l + (size_t)r * HEADS * DH, attn_r + (size_t)r * HEADS * DH,
        el, er, NH);
    const int* sr = src + (size_t)r * E;
    const int* dr = dst + (size_t)r * E;
    edge_score_max<<<blk(EH, B), B, 0, stream>>>(sr, dr, el, er, mkey, aexp, EH);
    edge_exp_sum<<<blk(EH, B), B, 0, stream>>>(dr, mkey, aexp, z, EH);
    edge_aggregate<<<blk(E * 32, B), B, 0, stream>>>(sr, dr, aexp, z, feat, agg, E);
    accum_out<<<blk(NC, B), B, 0, stream>>>(agg, bias + (size_t)r * IN_F, out, NC);
  }
  final_relu<<<blk(NC, B), B, 0, stream>>>(out, NC);
}
